// LabelAttentionLayer_49632642072960
// MI455X (gfx1250) — matrix operand order repaired and checked by reading, not compiled, not run on hardware
//
#include <hip/hip_runtime.h>
#include <stdint.h>

#define BB 16
#define SS 512
#define DD 512
#define LL 8000

typedef unsigned short u16;
typedef unsigned int   u32;
typedef unsigned long long u64;

typedef __bf16 bf16_t;
typedef bf16_t v16bf __attribute__((ext_vector_type(16)));
typedef float  v8f   __attribute__((ext_vector_type(8)));
typedef u32    v4u   __attribute__((ext_vector_type(4)));
typedef int    v8i   __attribute__((ext_vector_type(8)));
typedef int    v4i   __attribute__((ext_vector_type(4)));

struct B32x8 { uint4 lo; uint4 hi; };

// ---- helpers ---------------------------------------------------------------

static __device__ __forceinline__ u32 f2bf(float f) {
    // round-to-nearest-even f32 -> bf16 (payload in low 16 bits)
    u32 u = __float_as_uint(f);
    u32 r = u + 0x7FFFu + ((u >> 16) & 1u);
    return r >> 16;
}

// Load a 16-element bf16 fragment as two 16-byte chunks (lo at p, hi at p+gap ushorts).
static __device__ __forceinline__ v16bf load_frag(const u16* p, int gap) {
    B32x8 u;
    u.lo = *reinterpret_cast<const uint4*>(p);
    u.hi = *reinterpret_cast<const uint4*>(p + gap);
    return __builtin_bit_cast(v16bf, u);
}

static __device__ __forceinline__ v8f wmma_bf16(v16bf a, v16bf b, v8f c) {
    // D = A(16x32 bf16) x B(32x16 bf16) + C(16x16 f32)
    return __builtin_amdgcn_wmma_f32_16x16x32_bf16(false, a, false, b, (short)0, c,
                                                   false, false);
}

// Async copy 16B global -> LDS (per-lane), tracked with ASYNCcnt.
static __device__ __forceinline__ void async_copy_b128(u32 lds_off, const void* gptr) {
    u64 ga = (u64)(uintptr_t)gptr;
    asm volatile("global_load_async_to_lds_b128 %0, %1, off"
                 :: "v"(lds_off), "v"(ga)
                 : "memory");
}

static __device__ __forceinline__ void wait_asynccnt0() {
    asm volatile("s_wait_asynccnt 0x0" ::: "memory");
}

// ---- stage 0: f32 -> bf16 conversion (8 elems / thread, 16B stores) --------

__global__ void cvt_kernel(const float* __restrict__ src, u16* __restrict__ dst, int n8) {
    int i = blockIdx.x * blockDim.x + threadIdx.x;
    if (i >= n8) return;
    const float4* s = reinterpret_cast<const float4*>(src) + (size_t)i * 2;
    float4 f0 = s[0];
    float4 f1 = s[1];
    uint4 o;
    o.x = f2bf(f0.x) | (f2bf(f0.y) << 16);
    o.y = f2bf(f0.z) | (f2bf(f0.w) << 16);
    o.z = f2bf(f1.x) | (f2bf(f1.y) << 16);
    o.w = f2bf(f1.z) | (f2bf(f1.w) << 16);
    reinterpret_cast<uint4*>(dst)[i] = o;
}

// ---- stage 1: per-(b,s) softmax stats over the full label axis -------------
// grid = B * (S/16); block = 256 (8 waves). Each workgroup owns 16 s-columns
// of one batch; waves split the 500 label tiles and keep an online (max, sum).

__global__ void __launch_bounds__(256) stats_kernel(const u16* __restrict__ lstm_bf,
                                                    const u16* __restrict__ lab_bf,
                                                    float* __restrict__ row_max,
                                                    float* __restrict__ row_rsum) {
    __shared__ u16  lds_lstm[16][DD + 8];  // [s_local][d]
    __shared__ float red_m[8][16];
    __shared__ float red_s[8][16];

    const int tid = threadIdx.x;
    const int b  = blockIdx.x >> 5;
    const int s0 = (blockIdx.x & 31) * 16;

    // asynchronously stage the 16 x 512 lstm slice (bf16) into LDS (ASYNCcnt path)
    for (int idx = tid; idx < 16 * 64; idx += 256) {
        int s = idx >> 6, dc = (idx & 63) * 8;
        const u16* g = lstm_bf + ((size_t)(b * SS + s0 + s)) * DD + dc;
        async_copy_b128((u32)(uintptr_t)&lds_lstm[s][dc], g);
    }
    wait_asynccnt0();
    __syncthreads();

    const int wave = tid >> 5, lane = tid & 31;
    const int nlo = lane & 15, hi = lane >> 4;

    float m = -__builtin_inff();
    float sum = 0.0f;

    for (int t = wave; t < LL / 16; t += 8) {
        const int l0 = t * 16;
        v8f c = {0.f, 0.f, 0.f, 0.f, 0.f, 0.f, 0.f, 0.f};
        // A row pointer: label row (lane&15), K-half offset hi*8 per ISA A layout
        const u16* arow = lab_bf + (size_t)(l0 + nlo) * DD + hi * 8;
        // speculative prefetch of this wave's next label tile (t+8 -> +128 rows)
        __builtin_prefetch(arow + (size_t)128 * DD, 0, 0);
#pragma unroll
        for (int kk = 0; kk < 16; ++kk) {
            v16bf a  = load_frag(arow + kk * 32, 16);
            // B: column n = lane&15, K offset hi*8, second chunk +16 (table order)
            v16bf bm = load_frag(&lds_lstm[nlo][kk * 32 + hi * 8], 16);
            c = wmma_bf16(a, bm, c);
        }
        // online softmax update: this lane holds 8 labels for s-column (s0+nlo)
        float tm = c[0];
#pragma unroll
        for (int r = 1; r < 8; ++r) tm = fmaxf(tm, c[r]);
        float mn = fmaxf(m, tm);
        float ss = 0.0f;
#pragma unroll
        for (int r = 0; r < 8; ++r) ss += __expf(c[r] - mn);
        sum = sum * __expf(m - mn) + ss;
        m = mn;
    }

    // merge the two lane-halves that cover the same s-column (M 0..7 vs 8..15)
    {
        float om = __shfl_xor(m, 16, 32);
        float os = __shfl_xor(sum, 16, 32);
        float mn = fmaxf(m, om);
        sum = sum * __expf(m - mn) + os * __expf(om - mn);
        m = mn;
    }

    if (lane < 16) { red_m[wave][lane] = m; red_s[wave][lane] = sum; }
    __syncthreads();

    if (tid < 16) {
        float M = red_m[0][tid], Sm = red_s[0][tid];
#pragma unroll
        for (int w = 1; w < 8; ++w) {
            float mw = red_m[w][tid], sw = red_s[w][tid];
            float mn = fmaxf(M, mw);
            Sm = Sm * __expf(M - mn) + sw * __expf(mw - mn);
            M = mn;
        }
        row_max[b * SS + s0 + tid]  = M;
        row_rsum[b * SS + s0 + tid] = 1.0f / Sm;
    }
}

// ---- stage 2: fused weight recompute + second GEMM -------------------------
// grid = B * (L/16); block = 256 (8 waves). Workgroup owns 16 labels; loops
// over 32-s tiles: waves 0-1 recompute the 16x32 score tile via WMMA, apply
// exp(score-m)*rsum, store bf16 weights in LDS; all 8 waves accumulate
// out(16l x 64d per wave) with a second WMMA contracting s (K=32).

__global__ void __launch_bounds__(256) out_kernel(const u16* __restrict__ lstm_bf,
                                                  const u16* __restrict__ lab_bf,
                                                  const float* __restrict__ row_max,
                                                  const float* __restrict__ row_rsum,
                                                  float* __restrict__ out) {
    __shared__ u16 lds_lab[16][DD];      // label tile [l][d] (dense: TDM-friendly)
    __shared__ u16 lds_dT[DD][40];       // lstm tile transposed [d][s(32)]
    __shared__ u16 lds_w[16][40];        // bf16 weight tile [l][s(32)]

    const int tid = threadIdx.x;
    const int b  = blockIdx.x / (LL / 16);
    const int l0 = (blockIdx.x % (LL / 16)) * 16;
    const int wave = tid >> 5, lane = tid & 31;
    const int nlo = lane & 15, hi = lane >> 4;

#if __has_builtin(__builtin_amdgcn_tensor_load_to_lds)
    // Stage the 16 x 512 bf16 label tile with one Tensor Data Mover descriptor:
    // 2D tile (dim0 = 512 elems of 2B, dim1 = 16 rows), stride = 512 (dense).
    if (wave == 0) {
        const u64 ga = (u64)(uintptr_t)(lab_bf + (size_t)l0 * DD);
        const u32 la = (u32)(uintptr_t)&lds_lab[0][0];
        v4u g0;
        g0.x = (u32)__builtin_amdgcn_readfirstlane(1);          // count=1, user D#
        g0.y = (u32)__builtin_amdgcn_readfirstlane((int)la);    // lds_addr
        g0.z = (u32)__builtin_amdgcn_readfirstlane((int)(u32)ga);        // global lo
        g0.w = (u32)__builtin_amdgcn_readfirstlane(
                   (int)((((u32)(ga >> 32)) & 0x1FFFFFFu) | (2u << 30))); // hi | type=2
        v8i g1;
        g1[0] = __builtin_amdgcn_readfirstlane(0x00010000);  // data_size=1 (2 bytes)
        g1[1] = __builtin_amdgcn_readfirstlane(0x02000000);  // tensor_dim0=512 (lo16)
        g1[2] = __builtin_amdgcn_readfirstlane(0x00100000);  // tensor_dim1=16  (lo16)
        g1[3] = __builtin_amdgcn_readfirstlane(0x02000000);  // tile_dim0=512
        g1[4] = __builtin_amdgcn_readfirstlane(16);          // tile_dim1=16
        g1[5] = __builtin_amdgcn_readfirstlane(512);         // tensor_dim0_stride
        g1[6] = __builtin_amdgcn_readfirstlane(0);
        g1[7] = __builtin_amdgcn_readfirstlane(0);
        v4i gz  = {0, 0, 0, 0};
        v8i gz8 = {0, 0, 0, 0, 0, 0, 0, 0};
        // 6-arg form (clang-23 / therock-10.0): groups 0..3, extra group, cpol
        __builtin_amdgcn_tensor_load_to_lds(g0, g1, gz, gz, gz8, 0);
        __builtin_amdgcn_s_wait_tensorcnt(0);
    }
#else
    for (int idx = tid; idx < 16 * 64; idx += 256) {
        int l = idx >> 6, dc = (idx & 63) * 8;
        uint4 v = *reinterpret_cast<const uint4*>(lab_bf + (size_t)(l0 + l) * DD + dc);
        *reinterpret_cast<uint4*>(&lds_lab[l][dc]) = v;
    }
#endif

    v8f acc[4];
#pragma unroll
    for (int n = 0; n < 4; ++n)
#pragma unroll
        for (int j = 0; j < 8; ++j) acc[n][j] = 0.0f;

    for (int st = 0; st < SS / 32; ++st) {
        const int s0 = st * 32;
        __syncthreads();  // previous iteration's LDS reads done (also covers lds_lab)

        // cooperatively load the 32 x 512 lstm tile, transposed into [d][s]
        for (int idx = tid; idx < 32 * 64; idx += 256) {
            int s = idx >> 6, dc = (idx & 63) * 8;
            uint4 v = *reinterpret_cast<const uint4*>(
                lstm_bf + ((size_t)(b * SS + s0 + s)) * DD + dc);
            lds_dT[dc + 0][s] = (u16)(v.x);
            lds_dT[dc + 1][s] = (u16)(v.x >> 16);
            lds_dT[dc + 2][s] = (u16)(v.y);
            lds_dT[dc + 3][s] = (u16)(v.y >> 16);
            lds_dT[dc + 4][s] = (u16)(v.z);
            lds_dT[dc + 5][s] = (u16)(v.z >> 16);
            lds_dT[dc + 6][s] = (u16)(v.w);
            lds_dT[dc + 7][s] = (u16)(v.w >> 16);
        }
        __syncthreads();

        if (wave < 2) {
            const int sl = wave * 16 + nlo;  // local s column this lane owns
            v8f c = {0.f, 0.f, 0.f, 0.f, 0.f, 0.f, 0.f, 0.f};
#pragma unroll
            for (int kk = 0; kk < 16; ++kk) {
                v16bf a = load_frag(&lds_lab[nlo][kk * 32 + hi * 16], 8);
                // gather B column (fixed s, 16 consecutive d) from [d][s] layout
                const int k0 = kk * 32 + hi * 16;
                u32 w[8];
#pragma unroll
                for (int j = 0; j < 8; ++j) {
                    u32 lo = lds_dT[k0 + 2 * j][sl];
                    u32 hw = lds_dT[k0 + 2 * j + 1][sl];
                    w[j] = lo | (hw << 16);
                }
                B32x8 u;
                u.lo = make_uint4(w[0], w[1], w[2], w[3]);
                u.hi = make_uint4(w[4], w[5], w[6], w[7]);
                v16bf bm = __builtin_bit_cast(v16bf, u);
                c = wmma_bf16(a, bm, c);
            }
            const int sg   = b * SS + s0 + sl;
            const float mc = row_max[sg];
            const float rc = row_rsum[sg];
#pragma unroll
            for (int r = 0; r < 8; ++r) {
                float wv = __expf(c[r] - mc) * rc;          // softmax weight
                lds_w[r + hi * 8][sl] = (u16)f2bf(wv);      // C row M = r + hi*8
            }
        }
        __syncthreads();

        // second GEMM: A = weights(16x32), B = lstmT(32 x 16) per d-chunk
        v16bf aw = load_frag(&lds_w[nlo][hi * 8], 16);
#pragma unroll
        for (int n = 0; n < 4; ++n) {
            const int dl = wave * 64 + n * 16 + nlo;        // column d this lane owns
            v16bf bm = load_frag(&lds_dT[dl][hi * 8], 16);  // s = hi*8 + 0..7, 16 + hi*8 + 0..7
            acc[n] = wmma_bf16(aw, bm, acc[n]);
        }
    }

    // write out: fragment row M = r + hi*8 -> label l0+M, column = d
#pragma unroll
    for (int n = 0; n < 4; ++n) {
        const int dg = wave * 64 + n * 16 + nlo;
#pragma unroll
        for (int r = 0; r < 8; ++r) {
            const int l = l0 + r + hi * 8;
            out[((size_t)b * LL + l) * DD + dg] = acc[n][r];
        }
    }
}

// ---- launch ----------------------------------------------------------------

extern "C" void kernel_launch(void* const* d_in, const int* in_sizes, int n_in,
                              void* d_out, int out_size, void* d_ws, size_t ws_size,
                              hipStream_t stream) {
    const float* lstm_f = (const float*)d_in[0];  // (B,S,D) f32
    const float* lab_f  = (const float*)d_in[1];  // (L,D)   f32
    float* out = (float*)d_out;                   // (B,L,D) f32

    char* ws = (char*)d_ws;
    const size_t OFF_LSTM = 0;
    const size_t OFF_LAB  = OFF_LSTM + (size_t)BB * SS * DD * 2;  //  8,388,608
    const size_t OFF_MAX  = OFF_LAB  + (size_t)LL * DD * 2;       // 16,580,608
    const size_t OFF_RSUM = OFF_MAX  + (size_t)BB * SS * 4;       // 16,613,376
    u16*   lstm_bf  = (u16*)(ws + OFF_LSTM);
    u16*   lab_bf   = (u16*)(ws + OFF_LAB);
    float* row_max  = (float*)(ws + OFF_MAX);
    float* row_rsum = (float*)(ws + OFF_RSUM);

    const int n8_lstm = BB * SS * DD / 8;  // 524,288
    const int n8_lab  = LL * DD / 8;       // 512,000
    cvt_kernel<<<(n8_lstm + 255) / 256, 256, 0, stream>>>(lstm_f, lstm_bf, n8_lstm);
    cvt_kernel<<<(n8_lab + 255) / 256, 256, 0, stream>>>(lab_f, lab_bf, n8_lab);

    stats_kernel<<<BB * (SS / 16), 256, 0, stream>>>(lstm_bf, lab_bf, row_max, row_rsum);
    out_kernel<<<BB * (LL / 16), 256, 0, stream>>>(lstm_bf, lab_bf, row_max, row_rsum, out);
}
